// DotProductAttention_7533372637539
// MI455X (gfx1250) — matrix operand order repaired and checked by reading, not compiled, not run on hardware
//
#include <hip/hip_runtime.h>
#include <stdint.h>

// ---- fixed problem sizes (from reference) ----
#define BATCH 32
#define SQ    2048
#define SK    2048
#define DH    64      // head dim
#define BM    128     // Q rows per workgroup (16 per wave)
#define BN    64      // keys per inner iteration
#define NWAVE 8       // 256 threads / wave32
#define NTHR  (NWAVE * 32)
#define CVTTHR 128              // pre-pass block size
#define PADH  8                 // LDS row padding (halfs) = 16B
#define ROWH  (BN + PADH)       // 72 halfs per Vt/P row
#define KROWH (DH + PADH)       // 72 halfs per Kf row

typedef _Float16 half_t;
typedef __attribute__((ext_vector_type(16))) _Float16 v16h;
typedef __attribute__((ext_vector_type(8)))  _Float16 v8h;
typedef __attribute__((ext_vector_type(4)))  _Float16 v4h;
typedef __attribute__((ext_vector_type(8)))  float    v8f;

// Pointer types matching the async-LDS builtin's signature:
//   arg0: int4 in addrspace(1) (global), arg1: int4 in addrspace(3) (LDS)
typedef int v4i __attribute__((vector_size(16)));
typedef __attribute__((address_space(1))) v4i as1_v4i;
typedef __attribute__((address_space(3))) v4i as3_v4i;

// f16 workspace layout inside d_ws (16 MB needed):
//   Kh [BATCH][SK][DH]   row-major f16 keys
//   Vth[BATCH][DH][SK]   f16 values, pre-transposed (d major)
#define KH_ELEMS ((size_t)BATCH * SK * DH)

// ---------------------------------------------------------------------------
// CDNA5 WMMA f16 operand layouts (ISA 7.12.2, wave32):
//   A (16x32): lane row = lane%16;  e<8 -> k = 8*hi+e ; e>=8 -> k = 16+8*hi+(e-8)
//   B (32x16): lane col = lane%16;  e<8 -> k = 8*hi+e ; e>=8 -> k = 16+8*hi+(e-8)
//   C/D (16x16 f32): lane col N = lane%16; VGPR j -> row M = j + 8*hi
// ---------------------------------------------------------------------------
__device__ __forceinline__ v16h mk_a_f32s(const float* __restrict__ row, int hi, float scl) {
  const int o = hi ? 8 : 0;
  v16h r;
#pragma unroll
  for (int e = 0; e < 8; ++e) r[e] = (_Float16)(row[o + e] * scl);
#pragma unroll
  for (int e = 0; e < 8; ++e) r[8 + e] = (_Float16)(row[16 + o + e] * scl);
  return r;
}
__device__ __forceinline__ v16h mk_a_lds(const half_t* row, int hi) {
  const int o = hi ? 8 : 0;
  v16h r;
#pragma unroll
  for (int e = 0; e < 8; ++e) r[e] = row[o + e];
#pragma unroll
  for (int e = 0; e < 8; ++e) r[8 + e] = row[16 + o + e];
  return r;
}
__device__ __forceinline__ v16h mk_b_lds(const half_t* row, int hi) {
  const int o = hi ? 8 : 0;
  v16h r;
#pragma unroll
  for (int e = 0; e < 8; ++e) r[e] = row[o + e];
#pragma unroll
  for (int e = 0; e < 8; ++e) r[8 + e] = row[16 + o + e];
  return r;
}

#define WMMA_F16(A, Bm, C) \
  __builtin_amdgcn_wmma_f32_16x16x32_f16(false, (A), false, (Bm), (short)0, (C), false, false)

// ---- 16-lane max reduction on the VALU pipe via DPP16 ----------------------
template <int CTRL>
__device__ __forceinline__ float dpp_mv(float x) {
  return __int_as_float(
      __builtin_amdgcn_update_dpp(0, __float_as_int(x), CTRL, 0xF, 0xF, true));
}
__device__ __forceinline__ float red16_max(float v) {
  v = fmaxf(v, dpp_mv<0xB1>(v));   // quad_perm xor1
  v = fmaxf(v, dpp_mv<0x4E>(v));   // quad_perm xor2
  v = fmaxf(v, dpp_mv<0x141>(v));  // row_half_mirror
  v = fmaxf(v, dpp_mv<0x140>(v));  // row_mirror
  return v;
}

// ---- async global -> LDS copy (ASYNCcnt path), with sync fallback ----------
#if defined(__has_builtin)
#if __has_builtin(__builtin_amdgcn_global_load_async_to_lds_b128)
#define HAVE_ASYNC_LDS 1
#endif
#endif

__device__ __forceinline__ void async_cp16(const half_t* g, half_t* l) {
#ifdef HAVE_ASYNC_LDS
  __builtin_amdgcn_global_load_async_to_lds_b128(
      (as1_v4i*)(uintptr_t)g,
      (as3_v4i*)(uint32_t)(uintptr_t)l, 0, 0);
#else
  *(v8h*)l = *(const v8h*)g;   // synchronous fallback
#endif
}

#ifdef HAVE_ASYNC_LDS
#if defined(__has_builtin) && __has_builtin(__builtin_amdgcn_s_wait_asynccnt)
#define WAIT_ASYNC() __builtin_amdgcn_s_wait_asynccnt(0)
#else
#define WAIT_ASYNC() asm volatile("s_wait_asynccnt 0x0" ::: "memory")
#endif
#else
#define WAIT_ASYNC() ((void)0)
#endif

// ---------------------------------------------------------------------------
// Pre-pass: K f32 -> f16 row-major, V f32 -> f16 transposed Vth[b][d][k].
// One block per (batch, 64-key block); transpose goes through LDS so both
// the f32 reads and the f16 writes are coalesced.
// ---------------------------------------------------------------------------
__global__ __launch_bounds__(CVTTHR)
void convert_kv(const float* __restrict__ K, const float* __restrict__ V,
                half_t* __restrict__ Kh, half_t* __restrict__ Vth) {
  __shared__ half_t VtT[DH][ROWH];
  const int tid = threadIdx.x;
  const int b   = blockIdx.x >> 5;          // / (SK/BN)
  const int kb  = (blockIdx.x & 31) * BN;

  const float* Kb = K + ((size_t)b * SK + kb) * DH;
  const float* Vb = V + ((size_t)b * SK + kb) * DH;

  // K: straight f32 -> f16 convert, global -> global (v8h stores)
#pragma unroll
  for (int t = 0; t < 2; ++t) {
    const int task = tid + t * CVTTHR;      // 0..255
    const int kr = task >> 2;
    const int dd = (task & 3) * 16;
    const float* src = Kb + (size_t)kr * DH + dd;
    v8h a, c;
#pragma unroll
    for (int i = 0; i < 8; ++i) a[i] = (_Float16)src[i];
#pragma unroll
    for (int i = 0; i < 8; ++i) c[i] = (_Float16)src[8 + i];
    half_t* dst = Kh + ((size_t)b * SK + kb + kr) * DH + dd;
    *(v8h*)dst = a;
    *(v8h*)(dst + 8) = c;
  }

  // V: transpose 64x64 tile through LDS
  {
    const int kq = (tid & 15) * 4;
    const int d0 = (tid >> 4) * 8;
    const float* v0 = Vb + (size_t)kq * DH + d0;
#pragma unroll
    for (int i = 0; i < 8; ++i) {
      v4h p;
      p[0] = (_Float16)v0[i];
      p[1] = (_Float16)v0[DH + i];
      p[2] = (_Float16)v0[2 * DH + i];
      p[3] = (_Float16)v0[3 * DH + i];
      *(v4h*)&VtT[d0 + i][kq] = p;
    }
  }
  __syncthreads();
#pragma unroll
  for (int t = 0; t < 4; ++t) {
    const int idx = tid + t * CVTTHR;       // 0..511
    const int d = idx >> 3;
    const int c8 = (idx & 7) * 8;
    *(v8h*)&Vth[((size_t)b * DH + d) * SK + kb + c8] = *(const v8h*)&VtT[d][c8];
  }
}

// ---------------------------------------------------------------------------
// Issue async fill of one 64-key block into LDS (4 x b128 per thread).
// ---------------------------------------------------------------------------
__device__ __forceinline__ void fill_block(
    int kb, int tid, const half_t* __restrict__ Khb, const half_t* __restrict__ Vthb,
    half_t (&Kf)[BN][KROWH], half_t (&Vt)[DH][ROWH]) {
#pragma unroll
  for (int t = 0; t < 2; ++t) {
    const int idx = tid + t * NTHR;         // 0..511
    const int r = idx >> 3;
    const int c = (idx & 7) * 8;
    async_cp16(Khb + (size_t)(kb + r) * DH + c, &Kf[r][c]);
  }
#pragma unroll
  for (int t = 0; t < 2; ++t) {
    const int idx = tid + t * NTHR;
    const int d = idx >> 3;
    const int c = (idx & 7) * 8;
    async_cp16(Vthb + (size_t)d * SK + kb + c, &Vt[d][c]);
  }
}

// ---------------------------------------------------------------------------
// One 64-key block of compute: 8 WMMA (S = Q K^T), online softmax (DPP max,
// exp2), 8 WMMA (O += P V) + 2 WMMA (l += P * ones).
// ---------------------------------------------------------------------------
template <bool MASK>
__device__ __forceinline__ void compute_block(
    int kb, int vl, int wv, int hi, int lo,
    const v16h aq0, const v16h aq1, const v16h bones,
    v8f o[4], v8f& lsum, float m_j[8],
    half_t (&Kf)[BN][KROWH], half_t (&Vt)[DH][ROWH],
    half_t (&Pbuf)[NWAVE][16][ROWH]) {
  const float MASKED = -1.4426950e6f;  // -1e6 * log2(e): exp2 -> exactly 0

  // ---- S = Q K^T : 4 tiles of 16 keys (scale pre-folded into aq)
  v8f s[4];
#pragma unroll
  for (int t = 0; t < 4; ++t) {
    const half_t* krow = &Kf[t * 16 + lo][0];
    v8f acc = {};
    acc = WMMA_F16(aq0, mk_b_lds(krow, hi), acc);
    acc = WMMA_F16(aq1, mk_b_lds(krow + 32, hi), acc);
    s[t] = acc;
  }

  if (MASK) {
#pragma unroll
    for (int t = 0; t < 4; ++t) {
      const bool ok = (kb + t * 16 + lo) < vl;
#pragma unroll
      for (int j = 0; j < 8; ++j) s[t][j] = ok ? s[t][j] : MASKED;
    }
  }

  // ---- online softmax (row max via DPP16; row sum comes from WMMA below)
#pragma unroll
  for (int j = 0; j < 8; ++j) {
    float mx = fmaxf(fmaxf(s[0][j], s[1][j]), fmaxf(s[2][j], s[3][j]));
    mx = red16_max(mx);
    const float mn = fmaxf(m_j[j], mx);
    const float al = __builtin_amdgcn_exp2f(m_j[j] - mn);
    m_j[j] = mn;
#pragma unroll
    for (int t = 0; t < 4; ++t) o[t][j] *= al;
    lsum[j] *= al;
    const int r = j + 8 * hi;  // C-tile row this lane owns
#pragma unroll
    for (int t = 0; t < 4; ++t)
      Pbuf[wv][r][t * 16 + lo] = (_Float16)__builtin_amdgcn_exp2f(s[t][j] - mn);
  }

  // P stores precede P loads in program order; DS is in-order per wave.
  asm volatile("" ::: "memory");

  // ---- O += P V ; l += P * ones (row sums, replicated across lanes)
  const v16h ap0 = mk_a_lds(&Pbuf[wv][lo][0], hi);
  const v16h ap1 = mk_a_lds(&Pbuf[wv][lo][32], hi);
#pragma unroll
  for (int td = 0; td < 4; ++td) {
    const half_t* vrow = &Vt[td * 16 + lo][0];
    o[td] = WMMA_F16(ap0, mk_b_lds(vrow, hi), o[td]);
    o[td] = WMMA_F16(ap1, mk_b_lds(vrow + 32, hi), o[td]);
  }
  lsum = WMMA_F16(ap0, bones, lsum);
  lsum = WMMA_F16(ap1, bones, lsum);
}

// ---------------------------------------------------------------------------
// Main flash-attention kernel: double-buffered async LDS pipeline, 8 waves.
// ---------------------------------------------------------------------------
__global__ __launch_bounds__(NTHR)
void flash_attn_fwd(const float* __restrict__ Q, const int* __restrict__ vlen,
                    const half_t* __restrict__ Kh, const half_t* __restrict__ Vth,
                    float* __restrict__ out) {
  __shared__ half_t KfL[2][BN][KROWH];           // f16 K blocks (double buffer)
  __shared__ half_t VtL[2][DH][ROWH];            // f16 V^T blocks (double buffer)
  __shared__ half_t Pbuf[NWAVE][16][ROWH];       // per-wave P tiles

  const int tid  = threadIdx.x;
  const int lane = tid & 31;
  const int wv   = tid >> 5;
  const int hi   = lane >> 4;
  const int lo   = lane & 15;

  const int b    = blockIdx.x / (SQ / BM);
  const int qblk = blockIdx.x % (SQ / BM);
  const int q0   = qblk * BM + wv * 16;  // this wave's 16 Q rows

  const int vl = vlen[b];

  const half_t* Khb  = Kh  + (size_t)b * SK * DH;
  const half_t* Vthb = Vth + (size_t)b * DH * SK;

  // Q A-operands with 1/sqrt(d)*log2(e) folded into the f16 conversion.
  const float SCL = 0.18033688011112042f;
  const float* qrow = Q + ((size_t)b * SQ + q0 + lo) * DH;
  const v16h aq0 = mk_a_f32s(qrow, hi, SCL);
  const v16h aq1 = mk_a_f32s(qrow + 32, hi, SCL);

  v16h bones;
#pragma unroll
  for (int e = 0; e < 16; ++e) bones[e] = (_Float16)1.0f;

  v8f o[4] = {v8f{}, v8f{}, v8f{}, v8f{}};
  v8f lsum = {};
  float m_j[8];
#pragma unroll
  for (int j = 0; j < 8; ++j) m_j[j] = -3.0e38f;

  // vl is uniform per workgroup -> uniform pipeline control flow.
  const int NB     = (vl + BN - 1) / BN;  // >= 1 (vl >= 1)
  const int NBfull = vl / BN;

  fill_block(0, tid, Khb, Vthb, KfL[0], VtL[0]);
  int cur = 0;
  for (int ib = 0; ib < NBfull; ++ib) {
    WAIT_ASYNC();          // this wave's fill of buffer 'cur' has landed
    __syncthreads();       // => every wave's fill landed; prev reads done
    if (ib + 1 < NB)
      fill_block((ib + 1) * BN, tid, Khb, Vthb, KfL[cur ^ 1], VtL[cur ^ 1]);
    compute_block<false>(ib * BN, vl, wv, hi, lo, aq0, aq1, bones,
                         o, lsum, m_j, KfL[cur], VtL[cur], Pbuf);
    cur ^= 1;
  }
  if (NBfull < NB) {       // masked remainder block
    WAIT_ASYNC();
    __syncthreads();
    compute_block<true>(NBfull * BN, vl, wv, hi, lo, aq0, aq1, bones,
                        o, lsum, m_j, KfL[cur], VtL[cur], Pbuf);
  }

  // ---- epilogue: O / l, fp32 coalesced stores
#pragma unroll
  for (int j = 0; j < 8; ++j) {
    const int r = j + 8 * hi;
    const float inv = 1.0f / lsum[j];
    float* orow = out + ((size_t)b * SQ + q0 + r) * DH;
    orow[ 0 + lo] = o[0][j] * inv;
    orow[16 + lo] = o[1][j] * inv;
    orow[32 + lo] = o[2][j] * inv;
    orow[48 + lo] = o[3][j] * inv;
  }
}

extern "C" void kernel_launch(void* const* d_in, const int* in_sizes, int n_in,
                              void* d_out, int out_size, void* d_ws, size_t ws_size,
                              hipStream_t stream) {
  (void)in_sizes; (void)n_in; (void)out_size; (void)ws_size;  // needs >= 16 MB
  const float* Q    = (const float*)d_in[0];
  const float* K    = (const float*)d_in[1];
  const float* V    = (const float*)d_in[2];
  const int*   vlen = (const int*)d_in[3];
  float*       out  = (float*)d_out;

  half_t* Kh  = (half_t*)d_ws;
  half_t* Vth = Kh + KH_ELEMS;

  convert_kv<<<dim3(BATCH * (SK / BN)), dim3(CVTTHR), 0, stream>>>(K, V, Kh, Vth);
  flash_attn_fwd<<<dim3(BATCH * (SQ / BM)), dim3(NTHR), 0, stream>>>(Q, vlen, Kh, Vth, out);
}
